// BiMambaModel_4793183502531
// MI455X (gfx1250) — matrix operand order repaired and checked by reading, not compiled, not run on hardware
//
#include <hip/hip_runtime.h>
#include <hip/hip_bf16.h>
#include <math.h>
#include <stdint.h>

// ---------------- model constants ----------------
#define BB   16
#define LL   1024
#define DIMQ 272
#define EE   544      // E
#define E2   1088     // 2E
#define NNQ  256      // state N
#define RRQ  17       // R
#define MM   (BB*LL)  // 16384 tokens
#define KP1  288      // in_proj K padded (272 -> 288)

typedef _Float16 f16;
typedef __attribute__((ext_vector_type(16))) _Float16 v16h;
typedef __attribute__((ext_vector_type(8)))  _Float16 v8h;
typedef __attribute__((ext_vector_type(8)))  float    v8f;

// -------- CDNA5 async global->LDS copy (ASYNCcnt-tracked, no VGPR round trip) ----
__device__ __forceinline__ void async_ld_b32(const float* gptr, const float* lptr) {
  uint32_t lds = (uint32_t)(uintptr_t)lptr;          // low 32 bits = LDS offset
  uint64_t ga  = (uint64_t)(uintptr_t)gptr;
  asm volatile("global_load_async_to_lds_b32 %0, %1, off"
               :: "v"(lds), "v"(ga) : "memory");
}
__device__ __forceinline__ void wait_async0() {
  asm volatile("s_wait_asynccnt 0x0" ::: "memory");
}

// ================= WMMA GEMM: C[M,Nn] = A[M,K](f16) * W[Nn,K](f16)^T ==========
// per-wave tile: 32 rows x 64 cols, K multiple of 32.
// mode 0: store; mode 1: store softplus(acc + bias[col]); mode 2: C += acc
__global__ __launch_bounds__(256) void gemm_f16_wmma(
    const f16* __restrict__ A, const f16* __restrict__ W, float* __restrict__ C,
    int M, int Nn, int K, int lda, int ldw, int ldc,
    const float* __restrict__ bias, int mode)
{
  const int lane = threadIdx.x & 31;
  const int wid  = threadIdx.x >> 5;
  const int tilesN = (Nn + 63) >> 6;
  const int tilesM = M >> 5;
  int gw = blockIdx.x * 8 + wid;
  if (gw >= tilesM * tilesN) return;
  const int tm = gw / tilesN;
  const int tn = gw - tm * tilesN;
  const int m0 = tm << 5;
  const int n0 = tn << 6;
  const int g  = lane >> 4;    // half-wave group
  const int mr = lane & 15;

  v8f acc[2][4] = {};

  // A fragment (16x32 f16, MxK): lane g*16+mr holds row mr,
  //   halves 0..15 = K[k0 + g*16 .. +16) contiguous
  const f16* a0p = A + (size_t)(m0 + mr) * lda + g * 16;
  const f16* a1p = a0p + (size_t)16 * lda;
  // B fragment (32x16 f16, KxN): lane g*16+mr holds col mr, K[k0+g*16 .. +16) contiguous
  const f16* wp0 = W + (size_t)(n0 +  0 + mr) * ldw + g * 16;
  const f16* wp1 = W + (size_t)(n0 + 16 + mr) * ldw + g * 16;
  const f16* wp2 = W + (size_t)(n0 + 32 + mr) * ldw + g * 16;
  const f16* wp3 = W + (size_t)(n0 + 48 + mr) * ldw + g * 16;

  union U16 { v16h v; v8h p[2]; };

  for (int k0 = 0; k0 < K; k0 += 32) {
    U16 a0, a1, b0, b1, b2, b3;
    a0.p[0] = *(const v8h*)(a0p + k0);
    a0.p[1] = *(const v8h*)(a0p + k0 + 8);
    a1.p[0] = *(const v8h*)(a1p + k0);
    a1.p[1] = *(const v8h*)(a1p + k0 + 8);
    b0.p[0] = *(const v8h*)(wp0 + k0);  b0.p[1] = *(const v8h*)(wp0 + k0 + 8);
    b1.p[0] = *(const v8h*)(wp1 + k0);  b1.p[1] = *(const v8h*)(wp1 + k0 + 8);
    b2.p[0] = *(const v8h*)(wp2 + k0);  b2.p[1] = *(const v8h*)(wp2 + k0 + 8);
    b3.p[0] = *(const v8h*)(wp3 + k0);  b3.p[1] = *(const v8h*)(wp3 + k0 + 8);

    acc[0][0] = __builtin_amdgcn_wmma_f32_16x16x32_f16(false, a0.v, false, b0.v, (short)0, acc[0][0], false, false);
    acc[0][1] = __builtin_amdgcn_wmma_f32_16x16x32_f16(false, a0.v, false, b1.v, (short)0, acc[0][1], false, false);
    acc[0][2] = __builtin_amdgcn_wmma_f32_16x16x32_f16(false, a0.v, false, b2.v, (short)0, acc[0][2], false, false);
    acc[0][3] = __builtin_amdgcn_wmma_f32_16x16x32_f16(false, a0.v, false, b3.v, (short)0, acc[0][3], false, false);
    acc[1][0] = __builtin_amdgcn_wmma_f32_16x16x32_f16(false, a1.v, false, b0.v, (short)0, acc[1][0], false, false);
    acc[1][1] = __builtin_amdgcn_wmma_f32_16x16x32_f16(false, a1.v, false, b1.v, (short)0, acc[1][1], false, false);
    acc[1][2] = __builtin_amdgcn_wmma_f32_16x16x32_f16(false, a1.v, false, b2.v, (short)0, acc[1][2], false, false);
    acc[1][3] = __builtin_amdgcn_wmma_f32_16x16x32_f16(false, a1.v, false, b3.v, (short)0, acc[1][3], false, false);
  }

  // C/D layout: vgpr r = row (r + 8*g), col = mr within 16x16 tile
  #pragma unroll
  for (int i = 0; i < 2; ++i) {
    #pragma unroll
    for (int j = 0; j < 4; ++j) {
      int col = n0 + j * 16 + mr;
      if (col < Nn) {
        #pragma unroll
        for (int r = 0; r < 8; ++r) {
          int row = m0 + i * 16 + g * 8 + r;
          float v = acc[i][j][r];
          size_t idx = (size_t)row * ldc + col;
          if (mode == 0) {
            C[idx] = v;
          } else if (mode == 1) {
            float t = v + bias[col];
            C[idx] = (t > 20.f) ? t : log1pf(__expf(t));
          } else {
            C[idx] += v;
          }
        }
      }
    }
  }
}

// ============ LayerNorm over DIM=272 (one wave per row), optional residual add ====
__global__ __launch_bounds__(256) void ln_kernel(
    const float* __restrict__ src, const float* __restrict__ w, const float* __restrict__ b,
    const float* __restrict__ add, float* __restrict__ out, int rows)
{
  int row  = blockIdx.x * 8 + (threadIdx.x >> 5);
  int lane = threadIdx.x & 31;
  if (row >= rows) return;
  const float* p = src + (size_t)row * DIMQ;
  float vals[9];
  int cnt = 0;
  float s = 0.f;
  for (int i = lane; i < DIMQ; i += 32) { vals[cnt] = p[i]; s += vals[cnt]; ++cnt; }
  for (int m = 16; m > 0; m >>= 1) s += __shfl_xor(s, m, 32);
  float mean = s * (1.f / (float)DIMQ);
  float vsum = 0.f;
  for (int k = 0; k < cnt; ++k) { float d = vals[k] - mean; vsum += d * d; }
  for (int m = 16; m > 0; m >>= 1) vsum += __shfl_xor(vsum, m, 32);
  float rs = rsqrtf(vsum * (1.f / (float)DIMQ) + 1e-5f);
  cnt = 0;
  for (int i = lane; i < DIMQ; i += 32) {
    float o = (vals[cnt] - mean) * rs * w[i] + b[i];
    if (add) o += add[(size_t)row * DIMQ + i];
    out[(size_t)row * DIMQ + i] = o;
    ++cnt;
  }
}

// ============ f32 -> f16 with zero padding (rows x cols -> padRows x padCols) ====
__global__ __launch_bounds__(256) void convert_pad(
    const float* __restrict__ src, f16* __restrict__ dst,
    int rows, int cols, int padRows, int padCols)
{
  size_t idx = (size_t)blockIdx.x * 256 + threadIdx.x;
  size_t tot = (size_t)padRows * padCols;
  if (idx >= tot) return;
  int r = (int)(idx / padCols);
  int c = (int)(idx - (size_t)r * padCols);
  float v = (r < rows && c < cols) ? src[(size_t)r * cols + c] : 0.f;
  dst[idx] = (f16)v;
}

// ============ depthwise causal conv (DCONV=4) + bias + SiLU, f16 out ============
__global__ __launch_bounds__(256) void conv_silu_kernel(
    const float* __restrict__ xz, const float* __restrict__ cw, const float* __restrict__ cb,
    f16* __restrict__ xi_h, int dir)
{
  size_t idx = (size_t)blockIdx.x * 256 + threadIdx.x;
  if (idx >= (size_t)MM * EE) return;
  int m = (int)(idx / EE);
  int e = (int)(idx - (size_t)m * EE);
  int bq = m / LL, l = m - bq * LL;
  float acc = cb[e];
  #pragma unroll
  for (int k = 0; k < 4; ++k) {
    int sl = (dir == 0) ? (l - 3 + k) : (l + 3 - k);
    if (sl >= 0 && sl < LL)
      acc += cw[e * 4 + k] * xz[(size_t)(bq * LL + sl) * E2 + e];
  }
  float s = acc / (1.f + __expf(-acc));   // SiLU
  xi_h[idx] = (f16)s;
}

// ============ pack dt (prj cols [0,17)) into f16 M x 32, zero padded =============
__global__ __launch_bounds__(256) void pack_dt_kernel(
    const float* __restrict__ prj, f16* __restrict__ dt_h)
{
  size_t idx = (size_t)blockIdx.x * 256 + threadIdx.x;
  if (idx >= (size_t)MM * 32) return;
  int m = (int)(idx >> 5);
  int j = (int)(idx & 31);
  float v = (j < RRQ) ? prj[(size_t)m * EE + j] : 0.f;
  dt_h[idx] = (f16)v;
}

// ============ selective scan =====================================================
// grid: BB*17 blocks (b, e-tile of 32). block: 256 thr = 8 waves.
// wave handles 4 e channels; 8 lanes per e, each lane owns 32 n-states in regs.
// Double-buffered LDS staging of B/C via CDNA5 async global->LDS loads: the
// next step's operands stream in (ASYNCcnt) while the current step's exp/fma
// chain runs; one barrier per timestep.
__global__ __launch_bounds__(256) void scan_kernel(
    const float* __restrict__ prj, const float* __restrict__ delta,
    const f16* __restrict__ xi_h, const float* __restrict__ A_log,
    const float* __restrict__ Dp, float* __restrict__ y, int dir)
{
  __shared__ float sB[2][NNQ];
  __shared__ float sC[2][NNQ];
  __shared__ float sD[2][32];
  __shared__ float sU[2][32];

  const int bx = blockIdx.x;
  const int bq = bx / 17;
  const int e0 = (bx - bq * 17) * 32;
  const int t  = threadIdx.x;
  const int lane = t & 31;
  const int wid  = t >> 5;
  const int eLocal = wid * 4 + (lane >> 3);
  const int e  = e0 + eLocal;
  const int c  = lane & 7;
  const int n0 = c * 32;

  float h[32];
  float Aj[32];
  #pragma unroll
  for (int j = 0; j < 32; ++j) {
    h[j]  = 0.f;
    Aj[j] = -__expf(A_log[(size_t)e * NNQ + n0 + j]);
  }
  const float dscale = Dp[e];

  // stage step tt2 into LDS buffer buf (async for the 2KB B/C block)
  auto stage = [&](int tt2, int buf) {
    if (tt2 >= LL) return;
    const int l2 = (dir == 0) ? tt2 : (LL - 1 - tt2);
    const size_t row2 = (size_t)bq * LL + l2;
    async_ld_b32(&prj[row2 * EE + RRQ + t],       &sB[buf][t]);
    async_ld_b32(&prj[row2 * EE + RRQ + NNQ + t], &sC[buf][t]);
    if (t < 32) {
      sD[buf][t] = delta[row2 * EE + e0 + t];
      sU[buf][t] = (float)xi_h[row2 * EE + e0 + t];
    }
    // pull PRJ lines 8 steps ahead from HBM into L2
    if (t < 16) {
      int ta = tt2 + 8; if (ta > LL - 1) ta = LL - 1;
      int la = (dir == 0) ? ta : (LL - 1 - ta);
      __builtin_prefetch(&prj[((size_t)bq * LL + la) * EE + RRQ + t * 32], 0, 3);
    }
  };

  stage(0, 0);
  wait_async0();
  __syncthreads();

  for (int tt = 0; tt < LL; ++tt) {
    const int cur = tt & 1;
    stage(tt + 1, cur ^ 1);                 // overlap next step's staging

    const int l   = (dir == 0) ? tt : (LL - 1 - tt);
    const size_t row = (size_t)bq * LL + l;
    const float d  = sD[cur][eLocal];
    const float u  = sU[cur][eLocal];
    const float du = d * u;
    float acc = 0.f;
    #pragma unroll
    for (int j = 0; j < 32; ++j) {
      float dec = __expf(d * Aj[j]);
      h[j] = dec * h[j] + du * sB[cur][n0 + j];
      acc += h[j] * sC[cur][n0 + j];
    }
    acc += __shfl_xor(acc, 1, 32);
    acc += __shfl_xor(acc, 2, 32);
    acc += __shfl_xor(acc, 4, 32);
    if (c == 0) y[row * EE + e] = acc + dscale * u;

    wait_async0();      // my async loads into buf cur^1 are in LDS
    __syncthreads();    // everyone done reading buf cur + staging buf cur^1
  }
}

// ============ gating: g = y * silu(res), f16 out ================================
__global__ __launch_bounds__(256) void gate_kernel(
    const float* __restrict__ y, const float* __restrict__ xz, f16* __restrict__ gate_h)
{
  size_t idx = (size_t)blockIdx.x * 256 + threadIdx.x;
  if (idx >= (size_t)MM * EE) return;
  int m = (int)(idx / EE);
  int e = (int)(idx - (size_t)m * EE);
  float r = xz[(size_t)m * E2 + EE + e];
  float sr = r / (1.f + __expf(-r));
  gate_h[idx] = (f16)(y[idx] * sr);
}

// ============ pooled mean over L ================================================
__global__ __launch_bounds__(256) void pool_kernel(const float* __restrict__ X, float* __restrict__ pooled)
{
  int idx = blockIdx.x * 256 + threadIdx.x;
  if (idx >= BB * DIMQ) return;
  int bq = idx / DIMQ, d = idx - bq * DIMQ;
  float s = 0.f;
  for (int l = 0; l < LL; ++l) s += X[((size_t)bq * LL + l) * DIMQ + d];
  pooled[idx] = s * (1.f / (float)LL);
}

// ============ scores = pooled @ reg_w^T + reg_b =================================
__global__ __launch_bounds__(64) void score_kernel(
    const float* __restrict__ pooled, const float* __restrict__ rw,
    const float* __restrict__ rb, float* __restrict__ scores)
{
  int t = threadIdx.x;
  if (t >= BB * 4) return;
  int bq = t >> 2, s = t & 3;
  float acc = rb[s];
  for (int d = 0; d < DIMQ; ++d) acc += pooled[bq * DIMQ + d] * rw[s * DIMQ + d];
  scores[t] = acc;
}

// ================================ host orchestration =============================
extern "C" void kernel_launch(void* const* d_in, const int* in_sizes, int n_in,
                              void* d_out, int out_size, void* d_ws, size_t ws_size,
                              hipStream_t stream)
{
  const float* x_in  = (const float*)d_in[0];
  const float* ln1w  = (const float*)d_in[1];
  const float* ln1b  = (const float*)d_in[2];
  const float* ln2w  = (const float*)d_in[3];
  const float* ln2b  = (const float*)d_in[4];
  const float* ipw   = (const float*)d_in[5];
  const float* cw    = (const float*)d_in[6];
  const float* cb    = (const float*)d_in[7];
  const float* xpw   = (const float*)d_in[8];
  const float* dtw   = (const float*)d_in[9];
  const float* dtb   = (const float*)d_in[10];
  const float* alog  = (const float*)d_in[11];
  const float* dpp   = (const float*)d_in[12];
  const float* opw   = (const float*)d_in[13];
  const float* regw  = (const float*)d_in[14];
  const float* regb  = (const float*)d_in[15];
  (void)in_sizes; (void)n_in; (void)out_size; (void)ws_size;

  // workspace layout
  size_t off = 0;
  auto alloc = [&](size_t bytes) -> void* {
    off = (off + 255) & ~(size_t)255;
    void* p = (char*)d_ws + off;
    off += bytes;
    return p;
  };
  float* X      = (float*)alloc((size_t)MM * DIMQ * 4);
  float* XN     = (float*)alloc((size_t)MM * DIMQ * 4);
  f16*   XN_H   = (f16*)  alloc((size_t)MM * KP1  * 2);
  float* YSUM   = (float*)alloc((size_t)MM * DIMQ * 4);
  float* XZ     = (float*)alloc((size_t)MM * E2   * 4);
  f16*   XI_H   = (f16*)  alloc((size_t)MM * EE   * 2);
  float* PRJ    = (float*)alloc((size_t)MM * EE   * 4);   // 529 used, ld=544
  f16*   DT_H   = (f16*)  alloc((size_t)MM * 32   * 2);
  float* DELTA  = (float*)alloc((size_t)MM * EE   * 4);
  float* Y      = (float*)alloc((size_t)MM * EE   * 4);
  f16*   GATE_H = (f16*)  alloc((size_t)MM * EE   * 2);
  f16*   W_H    = (f16*)  alloc((size_t)1088 * 576 * 2);  // max padded weight

  auto gemm = [&](const f16* A, const f16* W, float* C, int M, int Nn, int K,
                  int lda, int ldw, int ldc, const float* bias, int mode) {
    int tilesN = (Nn + 63) / 64;
    int waves  = (M / 32) * tilesN;
    int blocks = (waves + 7) / 8;
    gemm_f16_wmma<<<blocks, 256, 0, stream>>>(A, W, C, M, Nn, K, lda, ldw, ldc, bias, mode);
  };
  auto cvt = [&](const float* src, f16* dst, int rows, int cols, int pr, int pc) {
    size_t tot = (size_t)pr * pc;
    convert_pad<<<(int)((tot + 255) / 256), 256, 0, stream>>>(src, dst, rows, cols, pr, pc);
  };

  const int gME = (int)(((size_t)MM * EE + 255) / 256);

  for (int l = 0; l < 6; ++l) {
    const float* srcX = (l == 0) ? x_in : X;
    // ln1 -> XN, then f16 pad
    ln_kernel<<<MM / 8, 256, 0, stream>>>(srcX, ln1w + l * DIMQ, ln1b + l * DIMQ, nullptr, XN, MM);
    cvt(XN, XN_H, MM, DIMQ, MM, KP1);

    for (int dir = 0; dir < 2; ++dir) {
      const int ld2 = l * 2 + dir;
      // in_proj: xz = xn @ iw^T   (16384x288 * 1088x288^T)
      cvt(ipw + (size_t)ld2 * E2 * DIMQ, W_H, E2, DIMQ, E2, KP1);
      gemm(XN_H, W_H, XZ, MM, E2, KP1, KP1, KP1, E2, nullptr, 0);
      // depthwise conv + silu -> XI_H (f16)
      conv_silu_kernel<<<gME, 256, 0, stream>>>(XZ, cw + (size_t)ld2 * EE * 4,
                                                cb + (size_t)ld2 * EE, XI_H, dir);
      // x_proj: prj = xi @ xpw^T  (Nn=529 padded to 576)
      cvt(xpw + (size_t)ld2 * 529 * EE, W_H, 529, EE, 576, EE);
      gemm(XI_H, W_H, PRJ, MM, 529, EE, EE, EE, EE, nullptr, 0);
      // dt -> f16 (K=17 padded to 32); delta = softplus(dt @ dtw^T + dtb)
      pack_dt_kernel<<<(int)(((size_t)MM * 32 + 255) / 256), 256, 0, stream>>>(PRJ, DT_H);
      cvt(dtw + (size_t)ld2 * EE * RRQ, W_H, EE, RRQ, 576, 32);
      gemm(DT_H, W_H, DELTA, MM, EE, 32, 32, 32, EE, dtb + (size_t)ld2 * EE, 1);
      // selective scan
      scan_kernel<<<BB * 17, 256, 0, stream>>>(PRJ, DELTA, XI_H,
                                               alog + (size_t)ld2 * EE * NNQ,
                                               dpp + (size_t)ld2 * EE, Y, dir);
      // gate: g = y * silu(res)
      gate_kernel<<<gME, 256, 0, stream>>>(Y, XZ, GATE_H);
      // out_proj: ysum (+)= g @ ow^T  (Nn=272 padded to 320)
      cvt(opw + (size_t)ld2 * DIMQ * EE, W_H, DIMQ, EE, 320, EE);
      gemm(GATE_H, W_H, YSUM, MM, DIMQ, EE, EE, EE, DIMQ, nullptr, dir == 0 ? 0 : 2);
    }
    // x = ln2(y1+y2) + xn
    ln_kernel<<<MM / 8, 256, 0, stream>>>(YSUM, ln2w + l * DIMQ, ln2b + l * DIMQ, XN, X, MM);
  }

  // outputs: scores (B*S) then pooled (B*DIM)
  float* out    = (float*)d_out;
  float* pooled = out + BB * 4;
  pool_kernel<<<(BB * DIMQ + 255) / 256, 256, 0, stream>>>(X, pooled);
  score_kernel<<<1, 64, 0, stream>>>(pooled, regw, regb, out);
}
